// EnhancedCompressedLLM_40965398069663
// MI455X (gfx1250) — compile-verified, matrix operand order checked by reading, not run on hardware
//
#include <hip/hip_runtime.h>
#include <math.h>

// ---------------- model constants ----------------
static constexpr int BATCH = 2;
static constexpr int SEQ   = 1024;
static constexpr int NTOK  = BATCH * SEQ;   // 2048
static constexpr int DM    = 1024;
static constexpr int NH    = 16;
static constexpr int HD    = 64;
static constexpr int NLAY  = 8;
static constexpr int RA    = 32;
static constexpr int RF    = 64;
static constexpr int DFF   = 4096;
static constexpr int VOC   = 32000;

typedef __attribute__((ext_vector_type(16))) _Float16 v16h;
typedef __attribute__((ext_vector_type(8)))  float    v8f;

#define WMMA_F16(a, b, c) \
  __builtin_amdgcn_wmma_f32_16x16x32_f16(false, (a), false, (b), (short)0, (c), false, false)

// ---------------- WMMA fragment helpers (16-bit A/B 16x32 layout, ISA 7.12.2) ----------------
// Lane's 16 K-values live in two contiguous 8-float runs: [ofs, ofs+8) and
// [16+ofs, 16+ofs+8) with ofs = hi ? 8 : 0  ->  4 x b128 loads per fragment.
__device__ __forceinline__ v16h frag_contig(const float* __restrict__ p, int hi, float scale) {
  const int ofs = hi ? 8 : 0;
  const float4 r0 = *(const float4*)(p + ofs);
  const float4 r1 = *(const float4*)(p + ofs + 4);
  const float4 r2 = *(const float4*)(p + 16 + ofs);
  const float4 r3 = *(const float4*)(p + 20 + ofs);
  v16h f;
  f[0]  = (_Float16)(r0.x * scale); f[1]  = (_Float16)(r0.y * scale);
  f[2]  = (_Float16)(r0.z * scale); f[3]  = (_Float16)(r0.w * scale);
  f[4]  = (_Float16)(r1.x * scale); f[5]  = (_Float16)(r1.y * scale);
  f[6]  = (_Float16)(r1.z * scale); f[7]  = (_Float16)(r1.w * scale);
  f[8]  = (_Float16)(r2.x * scale); f[9]  = (_Float16)(r2.y * scale);
  f[10] = (_Float16)(r2.z * scale); f[11] = (_Float16)(r2.w * scale);
  f[12] = (_Float16)(r3.x * scale); f[13] = (_Float16)(r3.y * scale);
  f[14] = (_Float16)(r3.z * scale); f[15] = (_Float16)(r3.w * scale);
  return f;
}

__device__ __forceinline__ int kgrp(int v, int hi) {
  return (v < 4) ? (2 * v + (hi ? 8 : 0)) : (2 * v + (hi ? 16 : 8));
}

// K values strided by ld (B column from row-major [K,N] source)
__device__ __forceinline__ v16h frag_strided(const float* __restrict__ p, int ld, int hi) {
  v16h f;
#pragma unroll
  for (int v = 0; v < 8; ++v) {
    int k = kgrp(v, hi);
    f[2 * v]     = (_Float16)p[k * ld];
    f[2 * v + 1] = (_Float16)p[(k + 1) * ld];
  }
  return f;
}

// ---------------- embedding gather (float4) ----------------
__global__ void embed_kernel(const int* __restrict__ ids, const float* __restrict__ emb,
                             float* __restrict__ X) {
  const int row = blockIdx.x;
  const int tid = threadIdx.x;                 // 256 threads, D/4 float4s
  const long id = (long)ids[row];
  ((float4*)X)[(long)row * (DM / 4) + tid] = ((const float4*)emb)[id * (DM / 4) + tid];
}

// ---------------- layernorm (one block of 256 per row, float4 I/O) ----------------
__global__ void ln_kernel(const float* __restrict__ x, const float* __restrict__ w,
                          const float* __restrict__ b, float* __restrict__ out) {
  __shared__ float red[16];
  const int row = blockIdx.x;
  const int tid = threadIdx.x;
  const float4 v = ((const float4*)x)[(long)row * (DM / 4) + tid];
  float s  = v.x + v.y + v.z + v.w;
  float ss = v.x * v.x + v.y * v.y + v.z * v.z + v.w * v.w;
#pragma unroll
  for (int m = 16; m >= 1; m >>= 1) {
    s  += __shfl_xor(s, m, 32);
    ss += __shfl_xor(ss, m, 32);
  }
  if ((tid & 31) == 0) { red[tid >> 5] = s; red[8 + (tid >> 5)] = ss; }
  __syncthreads();
  s = 0.f; ss = 0.f;
#pragma unroll
  for (int i = 0; i < 8; ++i) { s += red[i]; ss += red[8 + i]; }
  const float mu  = s * (1.0f / DM);
  const float var = ss * (1.0f / DM) - mu * mu;
  const float inv = rsqrtf(var + 1e-5f);
  const float4 wv = ((const float4*)w)[tid];
  const float4 bv = ((const float4*)b)[tid];
  float4 o;
  o.x = (v.x - mu) * inv * wv.x + bv.x;
  o.y = (v.y - mu) * inv * wv.y + bv.y;
  o.z = (v.z - mu) * inv * wv.z + bv.z;
  o.w = (v.w - mu) * inv * wv.w + bv.w;
  ((float4*)out)[(long)row * (DM / 4) + tid] = o;
}

// ---------------- low-rank stage 1: T[N,r] = (X[N,K] @ W^T) * S  (W row-major [r,K]) ----------------
__global__ __launch_bounds__(32)
void lr_stage1(const float* __restrict__ X, const float* __restrict__ W,
               const float* __restrict__ Sv, float* __restrict__ T, int K, int r) {
  const int lane = threadIdx.x & 31;
  const int col  = lane & 15;
  const int hi   = lane >> 4;
  const int m0 = blockIdx.x * 16, n0 = blockIdx.y * 16;
  const float* arow = X + (long)(m0 + col) * K;
  const float* brow = W + (long)(n0 + col) * K;
  v8f c = {};
  for (int k0 = 0; k0 < K; k0 += 32) {
    if (k0 + 32 < K) {                     // uniform branch: EXEC untouched
      __builtin_prefetch(arow + k0 + 32, 0, 1);
      __builtin_prefetch(brow + k0 + 32, 0, 1);
    }
    v16h a = frag_contig(arow + k0, hi, 1.0f);
    v16h b = frag_contig(brow + k0, hi, 1.0f);
    c = WMMA_F16(a, b, c);
  }
  const float s = Sv[n0 + col];
#pragma unroll
  for (int i = 0; i < 8; ++i)
    T[(long)(m0 + i + 8 * hi) * r + n0 + col] = c[i] * s;
}

// ---------------- low-rank stage 2: Z[N,Dout] = T[N,R] @ U^T (+epilogue). U row-major [Dout,R]
// EPI: 0 = none, 1 = +bias, 2 = gelu(+bias), 3 = +bias+residual.
// One wave covers NT consecutive 16-wide N tiles; A fragments hoisted. ----------------
template <int EPI, int R, int NT>
__global__ __launch_bounds__(32)
void lr_stage2(const float* __restrict__ T, const float* __restrict__ U,
               const float* __restrict__ bias, const float* __restrict__ resid,
               float* __restrict__ Z, int Dout) {
  const int lane = threadIdx.x & 31;
  const int col  = lane & 15;
  const int hi   = lane >> 4;
  const int m0    = blockIdx.x * 16;
  const int nbase = blockIdx.y * NT * 16;
  const float* arow = T + (long)(m0 + col) * R;
  const v16h a0 = frag_contig(arow, hi, 1.0f);
  v16h a1;
  if (R == 64) a1 = frag_contig(arow + 32, hi, 1.0f);
#pragma unroll 1
  for (int t = 0; t < NT; ++t) {
    const int n0 = nbase + t * 16;
    const float* brow = U + (long)(n0 + col) * R;
    if (t + 1 < NT) __builtin_prefetch(brow + 16 * R, 0, 1);
    v8f c = {};
    c = WMMA_F16(a0, frag_contig(brow, hi, 1.0f), c);
    if (R == 64) c = WMMA_F16(a1, frag_contig(brow + 32, hi, 1.0f), c);
    const float bv = (EPI >= 1) ? bias[n0 + col] : 0.0f;
#pragma unroll
    for (int i = 0; i < 8; ++i) {
      const long idx = (long)(m0 + i + 8 * hi) * Dout + n0 + col;
      float y = c[i] + bv;
      if (EPI == 2) y = 0.5f * y * (1.0f + erff(y * 0.70710678118f));  // exact gelu
      if (EPI == 3) y += resid[idx];
      Z[idx] = y;
    }
  }
}

// ---------------- flash attention: one wave per (b, h, 16-query tile), full-seq softmax ----------------
__global__ __launch_bounds__(32)
void attn_kernel(const float* __restrict__ Qb, const float* __restrict__ Kb,
                 const float* __restrict__ Vb, float* __restrict__ Ab) {
  const int lane = threadIdx.x & 31;
  const int col  = lane & 15;
  const int hi   = lane >> 4;
  const int b  = blockIdx.z;
  const int h  = blockIdx.y;
  const int q0 = blockIdx.x * 16;
  const long base = (long)b * SEQ * DM + (long)h * HD;

  __shared__ float pbuf[16][36];  // 36-float stride: 16B-aligned rows, bank-spread

  // Q fragments (head dim 64 -> two 32-chunks), pre-scaled by 1/sqrt(HD)
  const float* Qp = Qb + base + (long)(q0 + col) * DM;
  const v16h aq0 = frag_contig(Qp, hi, 0.125f);
  const v16h aq1 = frag_contig(Qp + 32, hi, 0.125f);

  v8f o[4] = {v8f{}, v8f{}, v8f{}, v8f{}};
  float rmax[8], rsum[8];
#pragma unroll
  for (int i = 0; i < 8; ++i) { rmax[i] = -1e30f; rsum[i] = 0.f; }

  for (int kb = 0; kb < SEQ; kb += 32) {
    const float* K0 = Kb + base + (long)(kb + col) * DM;
    const float* K1 = K0 + 16 * DM;
    if (kb + 32 < SEQ) {                       // uniform: prefetch next K/V tiles
      __builtin_prefetch(K0 + 32 * DM, 0, 1);
      __builtin_prefetch(K1 + 32 * DM, 0, 1);
      __builtin_prefetch(Vb + base + (long)(kb + 32 + lane) * DM, 0, 1);
      __builtin_prefetch(Vb + base + (long)(kb + 32 + lane) * DM + 32, 0, 1);
    }
    // ---- scores for key columns kb..kb+15 (s0) and kb+16..kb+31 (s1)
    v8f s0 = {}, s1 = {};
    s0 = WMMA_F16(aq0, frag_contig(K0, hi, 1.0f), s0);
    s0 = WMMA_F16(aq1, frag_contig(K0 + 32, hi, 1.0f), s0);
    s1 = WMMA_F16(aq0, frag_contig(K1, hi, 1.0f), s1);
    s1 = WMMA_F16(aq1, frag_contig(K1 + 32, hi, 1.0f), s1);

    // ---- online softmax (row = i + 8*hi, spread over the 16-lane group)
#pragma unroll
    for (int i = 0; i < 8; ++i) {
      float m = fmaxf(s0[i], s1[i]);
      m = fmaxf(m, __shfl_xor(m, 1, 32));
      m = fmaxf(m, __shfl_xor(m, 2, 32));
      m = fmaxf(m, __shfl_xor(m, 4, 32));
      m = fmaxf(m, __shfl_xor(m, 8, 32));
      const float nm  = fmaxf(rmax[i], m);
      const float fac = __expf(rmax[i] - nm);
      rmax[i] = nm;
      const float p0 = __expf(s0[i] - nm);
      const float p1 = __expf(s1[i] - nm);
      float ls = p0 + p1;
      ls += __shfl_xor(ls, 1, 32);
      ls += __shfl_xor(ls, 2, 32);
      ls += __shfl_xor(ls, 4, 32);
      ls += __shfl_xor(ls, 8, 32);
      rsum[i] = rsum[i] * fac + ls;
#pragma unroll
      for (int nt = 0; nt < 4; ++nt) o[nt][i] *= fac;
      pbuf[i + 8 * hi][col]      = p0;   // relayout C -> A via LDS
      pbuf[i + 8 * hi][16 + col] = p1;
    }
    __syncthreads();

    // ---- P as A-fragment (row = lane&15): 4 x ds_load_b128 via shared helper
    const v16h pa = frag_contig(&pbuf[col][0], hi, 1.0f);

    // ---- O += P @ V (V rows kb..kb+31, 64 dims -> 4 column tiles)
    const float* Vp = Vb + base + (long)kb * DM;
#pragma unroll
    for (int nt = 0; nt < 4; ++nt)
      o[nt] = WMMA_F16(pa, frag_strided(Vp + nt * 16 + col, DM, hi), o[nt]);
    __syncthreads();
  }

  // ---- normalize and write merged heads
#pragma unroll
  for (int i = 0; i < 8; ++i) {
    const float inv = 1.0f / rsum[i];
    float* orow = Ab + base + (long)(q0 + i + 8 * hi) * DM;
#pragma unroll
    for (int nt = 0; nt < 4; ++nt) orow[nt * 16 + col] = o[nt][i] * inv;
  }
}

// ---------------- host orchestration ----------------
extern "C" void kernel_launch(void* const* d_in, const int* in_sizes, int n_in,
                              void* d_out, int out_size, void* d_ws, size_t ws_size,
                              hipStream_t stream) {
  (void)in_sizes; (void)n_in; (void)out_size; (void)ws_size;

  // setup_inputs() dict insertion order (recursive)
  const int*   ids   = (const int*)d_in[0];
  const float* emb   = (const float*)d_in[1];
  const float* ln1w  = (const float*)d_in[2];
  const float* ln1b  = (const float*)d_in[3];
  const float* ln2w  = (const float*)d_in[4];
  const float* ln2b  = (const float*)d_in[5];
  const float* qU = (const float*)d_in[6],  *qS = (const float*)d_in[7],
             * qV = (const float*)d_in[8],  *qb = (const float*)d_in[9];
  const float* kU = (const float*)d_in[10], *kS = (const float*)d_in[11],
             * kV = (const float*)d_in[12], *kb = (const float*)d_in[13];
  const float* vU = (const float*)d_in[14], *vS = (const float*)d_in[15],
             * vV = (const float*)d_in[16], *vb = (const float*)d_in[17];
  const float* oU = (const float*)d_in[18], *oS = (const float*)d_in[19],
             * oV = (const float*)d_in[20], *ob = (const float*)d_in[21];
  const float* upU = (const float*)d_in[22], *upS = (const float*)d_in[23],
             * upV = (const float*)d_in[24], *upb = (const float*)d_in[25];
  const float* dnU = (const float*)d_in[26], *dnS = (const float*)d_in[27],
             * dnV = (const float*)d_in[28], *dnb = (const float*)d_in[29];
  const float* lnfw = (const float*)d_in[30];
  const float* lnfb = (const float*)d_in[31];
  const float* hU = (const float*)d_in[32], *hS = (const float*)d_in[33],
             * hV = (const float*)d_in[34];

  // workspace layout
  const long ND = (long)NTOK * DM;
  float* X  = (float*)d_ws;
  float* Hb = X  + ND;
  float* Qb = Hb + ND;
  float* Kb = Qb + ND;
  float* Vb = Kb + ND;
  float* Ab = Vb + ND;
  float* Gb = Ab + ND;               // [NTOK, DFF]
  float* Tb = Gb + (long)NTOK * DFF; // [NTOK, RF]

  constexpr int NT = 8;              // N tiles per wave in stage 2
  const dim3 w32(32);
  const dim3 gR32(NTOK / 16, RA / 16);          // rank-32 stage1
  const dim3 gR64(NTOK / 16, RF / 16);          // rank-64 stage1
  const dim3 gD  (NTOK / 16, DM / (16 * NT));   // Dout = 1024
  const dim3 gF  (NTOK / 16, DFF / (16 * NT));  // Dout = 4096
  const dim3 gVoc(NTOK / 16, VOC / (16 * NT));  // Dout = 32000
  const dim3 gAtt(SEQ / 16, NH, BATCH);

  embed_kernel<<<NTOK, 256, 0, stream>>>(ids, emb, X);

  for (int l = 0; l < NLAY; ++l) {
    // --- attention sub-block ---
    ln_kernel<<<NTOK, 256, 0, stream>>>(X, ln1w + (long)l * DM, ln1b + (long)l * DM, Hb);

    lr_stage1<<<gR32, w32, 0, stream>>>(Hb, qV + (long)l * RA * DM, qS + (long)l * RA, Tb, DM, RA);
    lr_stage2<1, RA, NT><<<gD, w32, 0, stream>>>(Tb, qU + (long)l * DM * RA, qb + (long)l * DM, nullptr, Qb, DM);

    lr_stage1<<<gR32, w32, 0, stream>>>(Hb, kV + (long)l * RA * DM, kS + (long)l * RA, Tb, DM, RA);
    lr_stage2<1, RA, NT><<<gD, w32, 0, stream>>>(Tb, kU + (long)l * DM * RA, kb + (long)l * DM, nullptr, Kb, DM);

    lr_stage1<<<gR32, w32, 0, stream>>>(Hb, vV + (long)l * RA * DM, vS + (long)l * RA, Tb, DM, RA);
    lr_stage2<1, RA, NT><<<gD, w32, 0, stream>>>(Tb, vU + (long)l * DM * RA, vb + (long)l * DM, nullptr, Vb, DM);

    attn_kernel<<<gAtt, w32, 0, stream>>>(Qb, Kb, Vb, Ab);

    lr_stage1<<<gR32, w32, 0, stream>>>(Ab, oV + (long)l * RA * DM, oS + (long)l * RA, Tb, DM, RA);
    lr_stage2<3, RA, NT><<<gD, w32, 0, stream>>>(Tb, oU + (long)l * DM * RA, ob + (long)l * DM, X, X, DM);

    // --- FFN sub-block ---
    ln_kernel<<<NTOK, 256, 0, stream>>>(X, ln2w + (long)l * DM, ln2b + (long)l * DM, Hb);

    lr_stage1<<<gR64, w32, 0, stream>>>(Hb, upV + (long)l * RF * DM, upS + (long)l * RF, Tb, DM, RF);
    lr_stage2<2, RF, NT><<<gF, w32, 0, stream>>>(Tb, upU + (long)l * DFF * RF, upb + (long)l * DFF, nullptr, Gb, DFF);

    lr_stage1<<<gR64, w32, 0, stream>>>(Gb, dnV + (long)l * RF * DFF, dnS + (long)l * RF, Tb, DFF, RF);
    lr_stage2<3, RF, NT><<<gD, w32, 0, stream>>>(Tb, dnU + (long)l * DM * RF, dnb + (long)l * DM, X, X, DM);
  }

  // --- final LN + low-rank LM head ---
  ln_kernel<<<NTOK, 256, 0, stream>>>(X, lnfw, lnfb, Hb);
  lr_stage1<<<gR64, w32, 0, stream>>>(Hb, hV, hS, Tb, DM, RF);
  lr_stage2<0, RF, NT><<<gVoc, w32, 0, stream>>>(Tb, hU, nullptr, nullptr, (float*)d_out, VOC);
}
